// MultiHeadAttention_41188736368737
// MI455X (gfx1250) — compile-verified, matrix operand order checked by reading, not run on hardware
//
#include <hip/hip_runtime.h>
#include <hip/hip_bf16.h>

// ---------------- CDNA5 WMMA types ----------------
typedef __bf16 bf16_t;
typedef __attribute__((ext_vector_type(16))) __bf16 v16bf;
typedef __attribute__((ext_vector_type(8)))  float  v8f;
typedef __attribute__((ext_vector_type(4)))  unsigned int v4u;
typedef __attribute__((ext_vector_type(8)))  int v8i;
typedef __attribute__((ext_vector_type(4)))  int v4i;

#define B_   2
#define L_   2048
#define D_   1024
#define H_   16
#define HD_  64
#define MAXREL 128

// LDS tile geometry for the big GEMMs (K-step 32, row pitch padded to 40 elems = 80B).
// TDM reproduces this pitch via pad_interval=3 (16 DWORDs) + pad_amount=3 (4 DWORDs).
#define APITCH 40
#define ATILE  (64 * APITCH)     // A: 64 rows
#define BTILE  (128 * APITCH)    // B: 128 rows

union FragBF { v16bf v; uint4 q[2]; };

// A/B fragment (16x32 bf16) from row-major global source (ld mult of 8, k0 mult of 32).
// ISA 7.12.2: lane holds row (lane&15); K chunks [half*8,+8) and [16+half*8,+8).
__device__ __forceinline__ v16bf load_frag_bf16(const bf16_t* p, int ld, int row0, int k0, int lane) {
    const int r = lane & 15, half = lane >> 4;
    const bf16_t* base = p + (size_t)(row0 + r) * ld + k0 + half * 8;
    FragBF f;
    f.q[0] = *(const uint4*)(base);
    f.q[1] = *(const uint4*)(base + 16);
    return f.v;
}

// Fragment from an LDS-staged 32-wide K slice (row pitch APITCH).
__device__ __forceinline__ v16bf load_frag_lds(const bf16_t* p, int row0, int lane) {
    const int r = lane & 15, half = lane >> 4;
    const bf16_t* base = p + (row0 + r) * APITCH + half * 8;
    FragBF f;
    f.q[0] = *(const uint4*)(base);
    f.q[1] = *(const uint4*)(base + 16);
    return f.v;
}

// Fragment sourced from fp32 (attention probabilities), converted on load.
__device__ __forceinline__ v16bf load_frag_f32(const float* p, int ld, int row0, int k0, int lane) {
    const int r = lane & 15, half = lane >> 4;
    const float* base = p + (size_t)(row0 + r) * ld + k0 + half * 8;
    float4 a0 = *(const float4*)(base);
    float4 a1 = *(const float4*)(base + 4);
    float4 b0 = *(const float4*)(base + 16);
    float4 b1 = *(const float4*)(base + 20);
    v16bf v;
    v[0] = (__bf16)a0.x; v[1] = (__bf16)a0.y; v[2]  = (__bf16)a0.z; v[3]  = (__bf16)a0.w;
    v[4] = (__bf16)a1.x; v[5] = (__bf16)a1.y; v[6]  = (__bf16)a1.z; v[7]  = (__bf16)a1.w;
    v[8] = (__bf16)b0.x; v[9] = (__bf16)b0.y; v[10] = (__bf16)b0.z; v[11] = (__bf16)b0.w;
    v[12]= (__bf16)b1.x; v[13]= (__bf16)b1.y; v[14] = (__bf16)b1.z; v[15] = (__bf16)b1.w;
    return v;
}

__device__ __forceinline__ v8f wmma_bf16(v16bf a, v16bf b, v8f c) {
    return __builtin_amdgcn_wmma_f32_16x16x32_bf16(false, a, false, b, (short)0, c, false, false);
}

// ---------------- CDNA5 async global->LDS copy (ASYNCcnt path) ----------------
__device__ __forceinline__ void async_ld_b128(uint32_t lds_byte, uint32_t gbyte, uint64_t sbase) {
    asm volatile("global_load_async_to_lds_b128 %0, %1, %2"
                 :: "v"(lds_byte), "v"(gbyte), "s"(sbase) : "memory");
}

// ---------------- CDNA5 Tensor Data Mover: 2D tile -> LDS (TENSORcnt path) ----------------
// Loads tile_rows x 32 bf16 elements starting at gaddr (row stride Kdim elems) into LDS
// at lds_byte, padding each 64B row by 16B so the LDS pitch is APITCH elements.
__device__ __forceinline__ void tdm_load_tile(uint32_t lds_byte, uint64_t gaddr,
                                              int tensor_rows, int Kdim, int tile_rows) {
    v4u g0;
    g0[0] = 1u;                                      // count=1, user descriptor
    g0[1] = lds_byte;                                // lds_addr (bytes)
    g0[2] = (unsigned int)gaddr;                     // global_addr[31:0]
    g0[3] = (unsigned int)(gaddr >> 32) | (2u << 30);// global_addr[56:32] | type=2
    v8i g1;
    g1[0] = (int)((1u << 16)         // data_size = 2 bytes
                | (1u << 20)         // pad_enable
                | (3u << 22)         // pad_interval: 16 DWORDs (one 64B row)
                | (3u << 25));       // pad_amount: 4 DWORDs (16B)
    g1[1] = (int)(((unsigned int)Kdim & 0xFFFFu) << 16);                 // tensor_dim0[15:0]
    g1[2] = (int)((((unsigned int)Kdim) >> 16)
                | (((unsigned int)tensor_rows & 0xFFFFu) << 16));        // dim0 hi | dim1 lo
    g1[3] = (int)((((unsigned int)tensor_rows) >> 16) | (32u << 16));    // dim1 hi | tile_dim0=32
    g1[4] = tile_rows;                                                   // tile_dim1 (tile_dim2=0)
    g1[5] = Kdim;                                                        // tensor_dim0_stride lo
    g1[6] = 0;                                                           // stride hi | dim1_stride
    g1[7] = 0;
    v4i z4 = (v4i)(0);
    v8i z8 = (v8i)(0);
    __builtin_amdgcn_tensor_load_to_lds(g0, g1, z4, z4, z8, 0);
}

// ---------------- double-buffered WMMA GEMM core: 64x128 block tile ----------------
// X row-major [*,Kdim] (A tile via async-to-LDS), Wt row-major [N,Kdim] (B tile via TDM).
// 8 waves (2m x 4n), wave tile 32x32.
__device__ __forceinline__ void gemm_core(const bf16_t* __restrict__ X,
                                          const bf16_t* __restrict__ Wt,
                                          int m0b, int n0b, int Nrows, int Kdim,
                                          bf16_t* As, bf16_t* Bs,
                                          v8f acc[2][2]) {
    const int tid  = threadIdx.x;
    const int lane = tid & 31, wave = tid >> 5;
    const int mw = (wave & 1) * 32;
    const int nw = (wave >> 1) * 32;
    const uint64_t xg = (uint64_t)(uintptr_t)X;
    const uint64_t wg = (uint64_t)(uintptr_t)Wt;

    // A tile: 64 rows x 4 b128 chunks = 256 chunks, 1 per thread (ASYNCcnt path)
    auto issueA = [&](int kk, int buf) {
        int row = tid >> 2, cc = tid & 3;
        uint32_t lds = (uint32_t)(uintptr_t)&As[buf * ATILE + row * APITCH + cc * 8];
        uint32_t gb  = (uint32_t)(((m0b + row) * Kdim + kk + cc * 8) * 2);
        async_ld_b128(lds, gb, xg);
    };
    // B tile: one TDM descriptor for the whole 128x32 slab (TENSORcnt path), wave 0 only
    auto issueB = [&](int kk, int buf) {
        uint32_t lds = (uint32_t)(uintptr_t)&Bs[buf * BTILE];
        uint64_t ga  = wg + (uint64_t)((size_t)((n0b) * Kdim + kk) * 2);
        tdm_load_tile(lds, ga, Nrows, Kdim, 128);
    };

    issueA(0, 0);
    if (wave == 0) issueB(0, 0);
    const int nk = Kdim / 32;
    for (int k = 0; k < nk; ++k) {
        const int buf = k & 1;
        if (k + 1 < nk) {
            issueA((k + 1) * 32, buf ^ 1);
            if (wave == 0) issueB((k + 1) * 32, buf ^ 1);
            asm volatile("s_wait_asynccnt 0x1" ::: "memory");      // stage k A done
            if (wave == 0) __builtin_amdgcn_s_wait_tensorcnt(1);   // stage k B done
        } else {
            asm volatile("s_wait_asynccnt 0x0" ::: "memory");
            if (wave == 0) __builtin_amdgcn_s_wait_tensorcnt(0);
        }
        __syncthreads();
        v16bf a0 = load_frag_lds(As + buf * ATILE, mw,      lane);
        v16bf a1 = load_frag_lds(As + buf * ATILE, mw + 16, lane);
        v16bf b0 = load_frag_lds(Bs + buf * BTILE, nw,      lane);
        v16bf b1 = load_frag_lds(Bs + buf * BTILE, nw + 16, lane);
        acc[0][0] = wmma_bf16(a0, b0, acc[0][0]);
        acc[0][1] = wmma_bf16(a0, b1, acc[0][1]);
        acc[1][0] = wmma_bf16(a1, b0, acc[1][0]);
        acc[1][1] = wmma_bf16(a1, b1, acc[1][1]);
        __syncthreads();
    }
}

// ---------------- elementwise casts ----------------
__global__ __launch_bounds__(256) void cast_bf16_kernel(const float* __restrict__ s,
                                                        bf16_t* __restrict__ d, int n) {
    int i = blockIdx.x * blockDim.x + threadIdx.x;
    if (i < n) d[i] = (bf16_t)s[i];
}

// Wt[n*K + k] = W[k*N + n]
__global__ __launch_bounds__(256) void transpose_cast_kernel(const float* __restrict__ W,
                                                             bf16_t* __restrict__ Wt, int K, int N) {
    int idx = blockIdx.x * blockDim.x + threadIdx.x;
    if (idx >= K * N) return;
    int n = idx / K, k = idx % K;
    Wt[idx] = (bf16_t)W[(size_t)k * N + n];
}

// ---------------- QKV projection: C = Xbf[4096,1024] x Wt + bias, bf16 out ----------------
// mode 0: dst[b][h][l][d]   (Q, K)      mode 1: dst[b][h][d][l]  (V transposed)
__global__ __launch_bounds__(256) void proj_kernel(const bf16_t* __restrict__ X,
                                                   const bf16_t* __restrict__ Wt,
                                                   const float* __restrict__ bias,
                                                   bf16_t* __restrict__ dst, int mode) {
    __shared__ __align__(16) bf16_t As[2 * ATILE];
    __shared__ __align__(16) bf16_t Bs[2 * BTILE];
    const int lane = threadIdx.x & 31, wave = threadIdx.x >> 5;
    const int m0 = blockIdx.y * 64 + (wave & 1) * 32;
    const int n0 = blockIdx.x * 128 + (wave >> 1) * 32;
    v8f acc[2][2] = {};
    gemm_core(X, Wt, blockIdx.y * 64, blockIdx.x * 128, D_, D_, As, Bs, acc);

    const int cl = lane & 15, half = lane >> 4;
    for (int mi = 0; mi < 2; ++mi)
    for (int ni = 0; ni < 2; ++ni)
    for (int r = 0; r < 8; ++r) {
        int gm = m0 + mi * 16 + r + half * 8;
        int gn = n0 + ni * 16 + cl;
        float val = acc[mi][ni][r] + bias[gn];
        int b = gm >> 11, l = gm & (L_ - 1);
        int h = gn >> 6,  d = gn & (HD_ - 1);
        if (mode == 0)
            dst[(((size_t)(b * H_ + h)) * L_ + l) * HD_ + d] = (bf16_t)val;
        else
            dst[(((size_t)(b * H_ + h)) * HD_ + d) * L_ + l] = (bf16_t)val;
    }
}

// ---------------- energy = Q K^T * scale + rel_bias -> attention buffer (fp32) ----------------
__global__ __launch_bounds__(256) void scores_kernel(const bf16_t* __restrict__ Qh,
                                                     const bf16_t* __restrict__ Kh,
                                                     const float* __restrict__ rel_bias,
                                                     float* __restrict__ attn) {
    const int bh = blockIdx.z;
    const int h  = bh & (H_ - 1);
    const bf16_t* Q = Qh + (size_t)bh * L_ * HD_;
    const bf16_t* K = Kh + (size_t)bh * L_ * HD_;
    float* S = attn + (size_t)bh * L_ * L_;

    const int lane = threadIdx.x & 31, wave = threadIdx.x >> 5;
    const int m0 = blockIdx.y * 64 + (wave & 1) * 32;
    const int n0 = blockIdx.x * 128 + (wave >> 1) * 32;
    v8f acc[2][2] = {};
    for (int kk = 0; kk < HD_; kk += 32) {
        v16bf a0 = load_frag_bf16(Q, HD_, m0,      kk, lane);
        v16bf a1 = load_frag_bf16(Q, HD_, m0 + 16, kk, lane);
        v16bf b0 = load_frag_bf16(K, HD_, n0,      kk, lane);
        v16bf b1 = load_frag_bf16(K, HD_, n0 + 16, kk, lane);
        acc[0][0] = wmma_bf16(a0, b0, acc[0][0]);
        acc[0][1] = wmma_bf16(a0, b1, acc[0][1]);
        acc[1][0] = wmma_bf16(a1, b0, acc[1][0]);
        acc[1][1] = wmma_bf16(a1, b1, acc[1][1]);
    }
    const float scale = 0.125f;  // 1/sqrt(64)
    const int cl = lane & 15, half = lane >> 4;
    for (int mi = 0; mi < 2; ++mi)
    for (int ni = 0; ni < 2; ++ni)
    for (int r = 0; r < 8; ++r) {
        int gq = m0 + mi * 16 + r + half * 8;
        int gk = n0 + ni * 16 + cl;
        int relp = gk - gq;
        relp = relp < -MAXREL ? -MAXREL : (relp > MAXREL ? MAXREL : relp);
        float val = acc[mi][ni][r] * scale + rel_bias[(relp + MAXREL) * H_ + h];
        S[(size_t)gq * L_ + gk] = val;
    }
}

// ---------------- row softmax over Lk=2048, in place ----------------
__global__ __launch_bounds__(256) void softmax_kernel(float* __restrict__ attn) {
    float* p = attn + (size_t)blockIdx.x * L_;
    __shared__ float redm[8];
    __shared__ float reds[8];
    const int tid = threadIdx.x;
    float vals[8];
    float m = -1e30f;
    #pragma unroll
    for (int i = 0; i < 8; ++i) { vals[i] = p[tid + i * 256]; m = fmaxf(m, vals[i]); }
    #pragma unroll
    for (int off = 16; off > 0; off >>= 1) m = fmaxf(m, __shfl_xor(m, off, 32));
    if ((tid & 31) == 0) redm[tid >> 5] = m;
    __syncthreads();
    float bm = redm[0];
    #pragma unroll
    for (int i = 1; i < 8; ++i) bm = fmaxf(bm, redm[i]);
    float s = 0.f;
    #pragma unroll
    for (int i = 0; i < 8; ++i) { vals[i] = __expf(vals[i] - bm); s += vals[i]; }
    #pragma unroll
    for (int off = 16; off > 0; off >>= 1) s += __shfl_xor(s, off, 32);
    if ((tid & 31) == 0) reds[tid >> 5] = s;
    __syncthreads();
    float tot = 0.f;
    #pragma unroll
    for (int i = 0; i < 8; ++i) tot += reds[i];
    const float inv = 1.0f / tot;
    #pragma unroll
    for (int i = 0; i < 8; ++i) p[tid + i * 256] = vals[i] * inv;
}

// ---------------- ctx = P (fp32, cast on load) x V  -> bf16 [4096,1024] ----------------
__global__ __launch_bounds__(128) void attnv_kernel(const float* __restrict__ attn,
                                                    const bf16_t* __restrict__ Vt,
                                                    bf16_t* __restrict__ ctx) {
    const int bh = blockIdx.z;
    const int b = bh >> 4, h = bh & (H_ - 1);
    const float*  P = attn + (size_t)bh * L_ * L_;
    const bf16_t* V = Vt   + (size_t)bh * HD_ * L_;   // [64, 2048]
    const int lane = threadIdx.x & 31, wave = threadIdx.x >> 5;
    const int m0 = blockIdx.y * 64 + (wave & 1) * 32;
    const int n0 = (wave >> 1) * 32;
    v8f acc[2][2] = {};
    for (int kk = 0; kk < L_; kk += 32) {
        v16bf a0 = load_frag_f32 (P, L_, m0,      kk, lane);
        v16bf a1 = load_frag_f32 (P, L_, m0 + 16, kk, lane);
        v16bf b0 = load_frag_bf16(V, L_, n0,      kk, lane);
        v16bf b1 = load_frag_bf16(V, L_, n0 + 16, kk, lane);
        acc[0][0] = wmma_bf16(a0, b0, acc[0][0]);
        acc[0][1] = wmma_bf16(a0, b1, acc[0][1]);
        acc[1][0] = wmma_bf16(a1, b0, acc[1][0]);
        acc[1][1] = wmma_bf16(a1, b1, acc[1][1]);
    }
    const int cl = lane & 15, half = lane >> 4;
    for (int mi = 0; mi < 2; ++mi)
    for (int ni = 0; ni < 2; ++ni)
    for (int r = 0; r < 8; ++r) {
        int gq = m0 + mi * 16 + r + half * 8;
        int gd = n0 + ni * 16 + cl;
        ctx[((size_t)(b * L_ + gq)) * D_ + h * HD_ + gd] = (bf16_t)acc[mi][ni][r];
    }
}

// ---------------- out = ctx x Wo + bo, fp32 ----------------
__global__ __launch_bounds__(256) void outproj_kernel(const bf16_t* __restrict__ X,
                                                      const bf16_t* __restrict__ Wt,
                                                      const float* __restrict__ bias,
                                                      float* __restrict__ dst) {
    __shared__ __align__(16) bf16_t As[2 * ATILE];
    __shared__ __align__(16) bf16_t Bs[2 * BTILE];
    const int lane = threadIdx.x & 31, wave = threadIdx.x >> 5;
    const int m0 = blockIdx.y * 64 + (wave & 1) * 32;
    const int n0 = blockIdx.x * 128 + (wave >> 1) * 32;
    v8f acc[2][2] = {};
    gemm_core(X, Wt, blockIdx.y * 64, blockIdx.x * 128, D_, D_, As, Bs, acc);

    const int cl = lane & 15, half = lane >> 4;
    for (int mi = 0; mi < 2; ++mi)
    for (int ni = 0; ni < 2; ++ni)
    for (int r = 0; r < 8; ++r) {
        int gm = m0 + mi * 16 + r + half * 8;
        int gn = n0 + ni * 16 + cl;
        dst[(size_t)gm * D_ + gn] = acc[mi][ni][r] + bias[gn];
    }
}

// ---------------- launcher ----------------
extern "C" void kernel_launch(void* const* d_in, const int* in_sizes, int n_in,
                              void* d_out, int out_size, void* d_ws, size_t ws_size,
                              hipStream_t stream) {
    const float* query = (const float*)d_in[0];
    const float* key_  = (const float*)d_in[1];
    const float* value = (const float*)d_in[2];
    const float* Wq = (const float*)d_in[3];  const float* bq = (const float*)d_in[4];
    const float* Wk = (const float*)d_in[5];  const float* bk = (const float*)d_in[6];
    const float* Wv = (const float*)d_in[7];  const float* bv = (const float*)d_in[8];
    const float* Wo = (const float*)d_in[9];  const float* bo = (const float*)d_in[10];
    const float* rel_bias = (const float*)d_in[11];

    float* out  = (float*)d_out;                       // [B, L, D]
    float* attn = out + (size_t)B_ * L_ * D_;          // [B, H, L, L]

    const int ME = B_ * L_ * D_;   // 4194304
    const int WE = D_ * D_;        // 1048576

    char* ws = (char*)d_ws;
    size_t off = 0;
    auto alloc = [&](size_t bytes) -> void* {
        void* p = ws + off;
        off += (bytes + 255) & ~(size_t)255;
        return p;
    };
    bf16_t* Xq  = (bf16_t*)alloc((size_t)ME * 2);
    bf16_t* Xk  = (bf16_t*)alloc((size_t)ME * 2);
    bf16_t* Xv  = (bf16_t*)alloc((size_t)ME * 2);
    bf16_t* Wqt = (bf16_t*)alloc((size_t)WE * 2);
    bf16_t* Wkt = (bf16_t*)alloc((size_t)WE * 2);
    bf16_t* Wvt = (bf16_t*)alloc((size_t)WE * 2);
    bf16_t* Wot = (bf16_t*)alloc((size_t)WE * 2);
    bf16_t* Qh  = (bf16_t*)alloc((size_t)ME * 2);      // [B,H,L,64]
    bf16_t* Kh  = (bf16_t*)alloc((size_t)ME * 2);      // [B,H,L,64]
    bf16_t* Vt  = (bf16_t*)alloc((size_t)ME * 2);      // [B,H,64,L]
    bf16_t* ctx = (bf16_t*)alloc((size_t)ME * 2);      // [B*L, D]

    cast_bf16_kernel<<<(ME + 255) / 256, 256, 0, stream>>>(query, Xq, ME);
    cast_bf16_kernel<<<(ME + 255) / 256, 256, 0, stream>>>(key_,  Xk, ME);
    cast_bf16_kernel<<<(ME + 255) / 256, 256, 0, stream>>>(value, Xv, ME);
    transpose_cast_kernel<<<(WE + 255) / 256, 256, 0, stream>>>(Wq, Wqt, D_, D_);
    transpose_cast_kernel<<<(WE + 255) / 256, 256, 0, stream>>>(Wk, Wkt, D_, D_);
    transpose_cast_kernel<<<(WE + 255) / 256, 256, 0, stream>>>(Wv, Wvt, D_, D_);
    transpose_cast_kernel<<<(WE + 255) / 256, 256, 0, stream>>>(Wo, Wot, D_, D_);

    dim3 gproj(D_ / 128, (B_ * L_) / 64);
    proj_kernel<<<gproj, 256, 0, stream>>>(Xq, Wqt, bq, Qh, 0);
    proj_kernel<<<gproj, 256, 0, stream>>>(Xk, Wkt, bk, Kh, 0);
    proj_kernel<<<gproj, 256, 0, stream>>>(Xv, Wvt, bv, Vt, 1);

    dim3 gsc(L_ / 128, L_ / 64, B_ * H_);
    scores_kernel<<<gsc, 256, 0, stream>>>(Qh, Kh, rel_bias, attn);

    softmax_kernel<<<B_ * H_ * L_, 256, 0, stream>>>(attn);

    dim3 gav(1, L_ / 64, B_ * H_);
    attnv_kernel<<<gav, 128, 0, stream>>>(attn, Vt, ctx);

    outproj_kernel<<<gproj, 256, 0, stream>>>(ctx, Wot, bo, out);
}
